// Head_49864570307212
// MI455X (gfx1250) — matrix operand order repaired and checked by reading, not compiled, not run on hardware
//
#include <hip/hip_runtime.h>

// ---------------------------------------------------------------------------
// Fused single-head attention for MI455X (gfx1250, wave32, WMMA).
//   Phase 0: convert Wq/Wk/Wv fp32 -> bf16 in workspace.
//   Phase 1: projection GEMMs via v_wmma_f32_16x16x32_bf16, software-pipelined
//            with a 2x-unrolled ping-pong (no register-rotation movs):
//            qh = (k @ Wq^T + bq) * (32^-0.5 * log2e)   [reference swap bug kept]
//            kh =  q @ Wk^T + bk
//            vhT = (v @ Wv^T + bv) stored TRANSPOSED [64][S] per batch
//   Phase 2: flash attention, one wave per 16 query rows, online softmax,
//            scores computed transposed (tokens as M) so softmax reductions
//            are in-lane + one ds_swizzle, and P^T feeds WMMA-B directly.
//            K-tiles ping-pong double-buffered (2x unrolled loop); V-tiles
//            issued before the softmax so their latency hides under the
//            score WMMAs + exp2 chain.
// ---------------------------------------------------------------------------

#define DEV __device__ __forceinline__

#define NB      4
#define SLEN    4096
#define DMODEL  2048
#define DHEAD   64
// scale = 1/sqrt(32), folded with log2(e) into qh so we can use v_exp_f32(exp2)
#define QSCALE  0.25507590651230795f   // (1/sqrt(32)) * 1.4426950408889634

typedef __attribute__((ext_vector_type(16))) __bf16 v16bf;
typedef __attribute__((ext_vector_type(8)))  float  v8f;

union BFTile { uint4 q[2]; unsigned u[8]; __bf16 h[16]; v16bf v; };

// pack 2 floats to packed-bf16 dword (lowered to v_cvt_pk_bf16_f32)
static DEV unsigned pk2bf(float lo, float hi) {
  union { __bf16 h[2]; unsigned u; } x;
  x.h[0] = (__bf16)lo; x.h[1] = (__bf16)hi;
  return x.u;
}
// swap lane halves (xor 16) within wave32: ds_swizzle SWAPX16 = 0x401F
static DEV float swzf(float x) {
  return __builtin_bit_cast(float,
      __builtin_amdgcn_ds_swizzle(__builtin_bit_cast(int, x), 0x401F));
}
static DEV unsigned swzu(unsigned x) {
  return (unsigned)__builtin_amdgcn_ds_swizzle((int)x, 0x401F);
}

// Load a 16x32 (A) or 32x16 (B) bf16 WMMA operand: two b128 loads per lane.
static DEV v16bf ld_tile(const __bf16* p, int o0, int o1) {
  BFTile t;
  t.q[0] = *(const uint4*)(p + o0);
  t.q[1] = *(const uint4*)(p + o1);
  return t.v;
}
// Same, but source data is fp32 (convert on the fly with native bf16 casts).
static DEV v16bf ld_tile_f32(const float* p, int o0, int o1) {
  float4 a = *(const float4*)(p + o0);
  float4 b = *(const float4*)(p + o0 + 4);
  float4 c = *(const float4*)(p + o1);
  float4 d = *(const float4*)(p + o1 + 4);
  BFTile t;
  t.u[0] = pk2bf(a.x, a.y); t.u[1] = pk2bf(a.z, a.w);
  t.u[2] = pk2bf(b.x, b.y); t.u[3] = pk2bf(b.z, b.w);
  t.u[4] = pk2bf(c.x, c.y); t.u[5] = pk2bf(c.z, c.w);
  t.u[6] = pk2bf(d.x, d.y); t.u[7] = pk2bf(d.z, d.w);
  return t.v;
}
static DEV v8f wmma_bf16(v16bf a, v16bf b, v8f c) {
  return __builtin_amdgcn_wmma_f32_16x16x32_bf16(
      /*neg_a=*/false, a, /*neg_b=*/false, b,
      /*c_mod=*/(short)0, c, /*reuse_a=*/false, /*reuse_b=*/false);
}

// ---------------------------------------------------------------------------
// Phase 0: weights fp32 -> bf16 (64*2048 each, stored back-to-back)
// ---------------------------------------------------------------------------
__global__ __launch_bounds__(256) void cvt_w_kernel(
    const float* __restrict__ Wq, const float* __restrict__ Wk,
    const float* __restrict__ Wv, __bf16* __restrict__ dst) {
  int i = blockIdx.x * 256 + threadIdx.x;        // grid 512 -> 131072 ids
  dst[i]          = (__bf16)Wq[i];
  dst[131072 + i] = (__bf16)Wk[i];
  dst[262144 + i] = (__bf16)Wv[i];
}

// ---------------------------------------------------------------------------
// Phase 1: projections. Grid (128, 3): blockIdx.y = proj (scalar). 8 waves per
// block -> 1024 row tiles of 16 rows over the flat [B*S] dimension.
// D(16x64) = X(16x2048) @ W^T; K-chunks of 32, 4 WMMAs per chunk; ping-pong
// buffers alternate roles across the two unrolled stages (no rotation movs).
// ---------------------------------------------------------------------------
__global__ __launch_bounds__(256) void proj_kernel(
    const float* __restrict__ qin, const float* __restrict__ kin,
    const float* __restrict__ vin, const __bf16* __restrict__ Wb,
    const float* __restrict__ bq, const float* __restrict__ bk,
    const float* __restrict__ bv,
    __bf16* __restrict__ qh, __bf16* __restrict__ kh,
    __bf16* __restrict__ vhT) {
  const int lane = threadIdx.x & 31;
  const int half = lane >> 4;
  const int lr   = lane & 15;
  const int proj = blockIdx.y;                       // scalar: 0..2
  const int tile = blockIdx.x * (blockDim.x >> 5) + (threadIdx.x >> 5);
  const int rowbase = tile * 16;                     // flat row over [B*S]

  // NOTE reference swap: qh uses input k, kh uses input q.
  const float* X = (proj == 0) ? kin : ((proj == 1) ? qin : vin);
  const __bf16* W = Wb + proj * 131072;
  const float* bias = (proj == 0) ? bq : ((proj == 1) ? bk : bv);

  v8f acc[4];
  #pragma unroll
  for (int i = 0; i < 4; ++i) acc[i] = (v8f){0,0,0,0,0,0,0,0};

  const float* xrow = X + (size_t)(rowbase + lr) * DMODEL;   // A row (M = lr)
  const __bf16* wrow = W + (size_t)lr * DMODEL;              // B row (N = lr)
  const int aoff = half * 8;    // A: lane<16 K{0..7,16..23}, lane>=16 K{8..15,24..31}
  const int boff = half * 8;    // B: lane<16 K{0..7,16..23}, lane>=16 K{8..15,24..31}

  // ---- prologue: load chunk 0 into buffer set 0 ----
  v16bf A0 = ld_tile_f32(xrow, aoff, aoff + 16);
  v16bf A1;
  v16bf B0[4], B1[4];
  #pragma unroll
  for (int dg = 0; dg < 4; ++dg)
    B0[dg] = ld_tile(wrow + dg * 16 * DMODEL, boff, boff + 16);

  for (int kc = 0; kc < DMODEL; kc += 64) {
    // -- stage 0: prefetch kc+32 into set 1, compute chunk kc with set 0 --
    const int k1 = kc + 32;                      // always < DMODEL
    A1 = ld_tile_f32(xrow, k1 + aoff, k1 + aoff + 16);
    #pragma unroll
    for (int dg = 0; dg < 4; ++dg)
      B1[dg] = ld_tile(wrow + dg * 16 * DMODEL, k1 + boff, k1 + boff + 16);
    #pragma unroll
    for (int dg = 0; dg < 4; ++dg) acc[dg] = wmma_bf16(A0, B0[dg], acc[dg]);

    // -- stage 1: prefetch kc+64 into set 0, compute chunk kc+32 with set 1 --
    const int k2 = (kc + 64) & (DMODEL - 1);     // wraps on last iter (discarded)
    A0 = ld_tile_f32(xrow, k2 + aoff, k2 + aoff + 16);
    #pragma unroll
    for (int dg = 0; dg < 4; ++dg)
      B0[dg] = ld_tile(wrow + dg * 16 * DMODEL, k2 + boff, k2 + boff + 16);
    #pragma unroll
    for (int dg = 0; dg < 4; ++dg) acc[dg] = wmma_bf16(A1, B1[dg], acc[dg]);
  }

  const float sc = (proj == 0) ? QSCALE : 1.0f;
  #pragma unroll
  for (int dg = 0; dg < 4; ++dg) {
    const int d = dg * 16 + lr;
    const float bval = bias[d];
    #pragma unroll
    for (int r = 0; r < 8; ++r) {
      const float val = (acc[dg][r] + bval) * sc;   // C tile: M=r+8*half, N=lr
      const int row = rowbase + r + 8 * half;
      if (proj == 0) {
        qh[(size_t)row * DHEAD + d] = (__bf16)val;
      } else if (proj == 1) {
        kh[(size_t)row * DHEAD + d] = (__bf16)val;
      } else {
        const int batch = row >> 12, srow = row & (SLEN - 1);
        vhT[((size_t)batch * DHEAD + d) * SLEN + srow] = (__bf16)val;
      }
    }
  }
}

// ---------------------------------------------------------------------------
// Phase 2: flash attention. One wave per 16 query rows (1024 waves).
// Scores transposed: S^T(32tok x 16q) = Kh(32x64) @ Qh^T(64x16)  (4 WMMAs)
// out^T(64 x 16q)  += VhT(64x32tok) @ P^T(32tok x 16q)           (4 WMMAs)
// ---------------------------------------------------------------------------
__global__ __launch_bounds__(256) void attn_kernel(
    const __bf16* __restrict__ qh, const __bf16* __restrict__ kh,
    const __bf16* __restrict__ vhT, float* __restrict__ out) {
  const int lane = threadIdx.x & 31;
  const int half = lane >> 4;
  const int lr   = lane & 15;
  const int wave  = blockIdx.x * (blockDim.x >> 5) + (threadIdx.x >> 5);
  const int batch = wave >> 8;
  const int qbase = (wave & 255) * 16;

  // Qh^T as WMMA-B, resident for whole loop: feats 0..31 and 32..63
  const __bf16* qrow = qh + ((size_t)batch * SLEN + qbase + lr) * DHEAD;
  const v16bf BQ0 = ld_tile(qrow,      half * 8,      half * 8 + 16);
  const v16bf BQ1 = ld_tile(qrow, 32 + half * 8, 48 + half * 8);

  const __bf16* kbase = kh  + (size_t)batch * SLEN * DHEAD;
  const __bf16* kr0   = kbase + (size_t)lr * DHEAD;          // token row lr
  const __bf16* kr1   = kbase + (size_t)(16 + lr) * DHEAD;   // token row 16+lr
  const __bf16* vbase = vhT + (size_t)batch * DHEAD * SLEN;
  const __bf16* vrow  = vbase + (size_t)lr * SLEN;           // feat row lr

  v8f acc[4];
  #pragma unroll
  for (int i = 0; i < 4; ++i) acc[i] = (v8f){0,0,0,0,0,0,0,0};
  float m = -1e30f, l = 0.0f;
  const v8f zero = (v8f){0,0,0,0,0,0,0,0};
  const int aoff = half * 8;

  // Per-chunk computation: V loads for this chunk, score WMMAs with the given
  // (already resident) K tiles, online softmax, P^T build, rescale + PV WMMAs.
  auto chunk = [&](int kb, v16bf K00, v16bf K01, v16bf K10, v16bf K11) {
    // --- V loads issued first; consumed only after softmax ---
    v16bf VA[4];
    #pragma unroll
    for (int fg = 0; fg < 4; ++fg)
      VA[fg] = ld_tile(vrow + fg * 16 * SLEN, kb + half * 16, kb + half * 16 + 8);

    // --- scores S^T: two 16x16 C tiles (tokens 0-15, 16-31) ---
    v8f s1 = wmma_bf16(K00, BQ0, zero); s1 = wmma_bf16(K01, BQ1, s1);
    v8f s2 = wmma_bf16(K10, BQ0, zero); s2 = wmma_bf16(K11, BQ1, s2);

    // --- online softmax over 32 tokens (per qrow = per lane col) ---
    float cmax = s1[0];
    #pragma unroll
    for (int r = 1; r < 8; ++r) cmax = fmaxf(cmax, s1[r]);
    #pragma unroll
    for (int r = 0; r < 8; ++r) cmax = fmaxf(cmax, s2[r]);
    cmax = fmaxf(cmax, swzf(cmax));           // combine lane halves
    const float mn = fmaxf(m, cmax);
    const float sf = __builtin_amdgcn_exp2f(m - mn);   // scores carry log2e
    float p1[8], p2[8], ls = 0.0f;
    #pragma unroll
    for (int r = 0; r < 8; ++r) { p1[r] = __builtin_amdgcn_exp2f(s1[r] - mn); ls += p1[r]; }
    #pragma unroll
    for (int r = 0; r < 8; ++r) { p2[r] = __builtin_amdgcn_exp2f(s2[r] - mn); ls += p2[r]; }
    ls += swzf(ls);
    l = l * sf + ls;
    m = mn;

    // --- build P^T (32tok x 16q) in WMMA-B layout, registers only ---
    // B k order used here (same as VA): lane<16 K{0..15}, lane>=16 K{16..31}; VGPR v holds K=2v,2v+1.
    // C tiles: lane<16 holds tokens r, lane>=16 tokens r+8 -> one swizzle.
    BFTile bp;
    #pragma unroll
    for (int v = 0; v < 4; ++v) {
      const unsigned pa = pk2bf(p1[2 * v], p1[2 * v + 1]);   // T1 tokens 2v,2v+1
      const unsigned pb = pk2bf(p2[2 * v], p2[2 * v + 1]);   // T2 tokens 2v,2v+1
      const unsigned xa = swzu(pa);
      const unsigned xb = swzu(pb);
      bp.u[v]     = half ? xb : pa;   // K = 2v   (lo) / 16+2v (hi)
      bp.u[v + 4] = half ? pb : xa;   // K = 8+2v (lo) / 24+2v (hi)
    }

    // --- rescale accumulator, then PV WMMAs ---
    #pragma unroll
    for (int fg = 0; fg < 4; ++fg) {
      #pragma unroll
      for (int r = 0; r < 8; ++r) acc[fg][r] *= sf;
    }
    #pragma unroll
    for (int fg = 0; fg < 4; ++fg) acc[fg] = wmma_bf16(VA[fg], bp.v, acc[fg]);
  };

  // ---- prologue: K tiles for chunk 0 into set A ----
  v16bf KA00 = ld_tile(kr0, aoff,      aoff + 16);
  v16bf KA01 = ld_tile(kr0, 32 + aoff, 48 + aoff);
  v16bf KA10 = ld_tile(kr1, aoff,      aoff + 16);
  v16bf KA11 = ld_tile(kr1, 32 + aoff, 48 + aoff);
  v16bf KB00, KB01, KB10, KB11;

  for (int kb = 0; kb < SLEN; kb += 64) {
    // -- stage 0: prefetch K(kb+32) into set B, compute chunk kb with set A --
    {
      const int knb = (kb + 32) * DHEAD;         // always < SLEN*DHEAD
      KB00 = ld_tile(kr0 + knb, aoff,      aoff + 16);
      KB01 = ld_tile(kr0 + knb, 32 + aoff, 48 + aoff);
      KB10 = ld_tile(kr1 + knb, aoff,      aoff + 16);
      KB11 = ld_tile(kr1 + knb, 32 + aoff, 48 + aoff);
      chunk(kb, KA00, KA01, KA10, KA11);
    }
    // -- stage 1: prefetch K(kb+64) into set A, compute chunk kb+32 with B --
    {
      const int knb = ((kb + 64) & (SLEN - 1)) * DHEAD;  // wraps on last iter
      KA00 = ld_tile(kr0 + knb, aoff,      aoff + 16);
      KA01 = ld_tile(kr0 + knb, 32 + aoff, 48 + aoff);
      KA10 = ld_tile(kr1 + knb, aoff,      aoff + 16);
      KA11 = ld_tile(kr1 + knb, 32 + aoff, 48 + aoff);
      chunk(kb + 32, KB00, KB01, KB10, KB11);
    }
  }

  // --- epilogue: divide by softmax sum, store out[B*S][64] fp32 ---
  const float inv = 1.0f / l;
  const size_t orow = ((size_t)batch * SLEN + qbase + lr) * DHEAD;  // col=qrow=lr
  #pragma unroll
  for (int fg = 0; fg < 4; ++fg) {
    #pragma unroll
    for (int r = 0; r < 8; ++r) {
      out[orow + fg * 16 + r + 8 * half] = acc[fg][r] * inv;
    }
  }
}

// ---------------------------------------------------------------------------
extern "C" void kernel_launch(void* const* d_in, const int* in_sizes, int n_in,
                              void* d_out, int out_size, void* d_ws, size_t ws_size,
                              hipStream_t stream) {
  const float* q  = (const float*)d_in[0];
  const float* k  = (const float*)d_in[1];
  const float* v  = (const float*)d_in[2];
  const float* Wq = (const float*)d_in[3];
  const float* bq = (const float*)d_in[4];
  const float* Wk = (const float*)d_in[5];
  const float* bk = (const float*)d_in[6];
  const float* Wv = (const float*)d_in[7];
  const float* bv = (const float*)d_in[8];
  float* out = (float*)d_out;

  // Workspace layout (bf16 elements): qh | kh | vhT | Wq_bf|Wk_bf|Wv_bf
  // = 3*1,048,576 + 3*131,072 elems = ~6.75 MB
  __bf16* qh  = (__bf16*)d_ws;
  __bf16* kh  = qh  + (size_t)NB * SLEN * DHEAD;
  __bf16* vhT = kh  + (size_t)NB * SLEN * DHEAD;
  __bf16* Wb  = vhT + (size_t)NB * SLEN * DHEAD;

  cvt_w_kernel<<<512, 256, 0, stream>>>(Wq, Wk, Wv, Wb);
  proj_kernel<<<dim3(128, 3), 256, 0, stream>>>(q, k, v, Wb, bq, bk, bv, qh, kh, vhT);
  attn_kernel<<<128, 256, 0, stream>>>(qh, kh, vhT, out);
}
